// GNN2_21311627722806
// MI455X (gfx1250) — compile-verified, matrix operand order checked by reading, not run on hardware
//
#include <hip/hip_runtime.h>

// ---------------------------------------------------------------------------
// CDNA5 (gfx1250) GNN pipeline, wave32 + v_wmma_f32_16x16x32_bf16 everywhere.
// HBM-bound on adjacency reads (7 x 256MB fp32) -> read adj exactly once per
// GEMM, convert fp32->bf16 in registers, keep all reused operands bf16 in L2.
// Big GEMM is double-buffered: B staged with GLOBAL_LOAD_ASYNC_TO_LDS_B128
// (ASYNCcnt), A register-prefetched while WMMAs run on the other buffer.
// ---------------------------------------------------------------------------

typedef __attribute__((ext_vector_type(16))) __bf16 v16bf;
typedef __attribute__((ext_vector_type(8)))  __bf16 v8bf;
typedef __attribute__((ext_vector_type(8)))  float  v8f;
typedef __attribute__((ext_vector_type(4)))  int    v4i;

union Frag { v16bf v; v8bf h[2]; };

#define WMMA_BF16(a, b, c) \
  __builtin_amdgcn_wmma_f32_16x16x32_bf16(false, (a), false, (b), (short)0, (c), false, false)

// --- CDNA5 async global->LDS copy (16B per lane) + ASYNCcnt wait -----------
static __device__ __forceinline__ void async_cp_b128(void* lds, const void* g) {
#if __has_builtin(__builtin_amdgcn_global_load_async_to_lds_b128)
  __builtin_amdgcn_global_load_async_to_lds_b128(
      (__attribute__((address_space(1))) v4i*)(g),
      (__attribute__((address_space(3))) v4i*)(lds), 0, 0);
#else
  unsigned lds_addr = (unsigned)(unsigned long long)(lds);
  asm volatile("global_load_async_to_lds_b128 %0, %1, off"
               :: "v"(lds_addr), "v"(g) : "memory");
#endif
}
static __device__ __forceinline__ void wait_async0() {
#if __has_builtin(__builtin_amdgcn_s_wait_asynccnt)
  __builtin_amdgcn_s_wait_asynccnt(0);
#else
  asm volatile("s_wait_asynccnt 0x0" ::: "memory");
#endif
}

static __device__ __forceinline__ v8bf pack8(float4 a, float4 b) {
  v8bf r;
  r[0] = (__bf16)a.x; r[1] = (__bf16)a.y; r[2] = (__bf16)a.z; r[3] = (__bf16)a.w;
  r[4] = (__bf16)b.x; r[5] = (__bf16)b.y; r[6] = (__bf16)b.z; r[7] = (__bf16)b.w;
  return r;
}

// ---------------------------------------------------------------------------
// z = relu(A[8192x8192](fp32) @ BT^T) with optional row L2-norm.
// BT is the 128x8192 (n-major) bf16 operand. Output Xout: [8192x128] bf16.
// WG = 256 threads (8 waves), tile = 32 rows x 128 cols (full N), Kchunk = 64.
// Double-buffered LDS pipeline: async B, register-prefetched A.
// ---------------------------------------------------------------------------
__global__ __launch_bounds__(256)
void k_gemm_adj(const float* __restrict__ A, const __bf16* __restrict__ BT,
                __bf16* __restrict__ Xout, int do_norm)
{
  __shared__ __align__(16) __bf16 Alds[2][32 * 64];    //  2 x  4 KB
  __shared__ __align__(16) __bf16 Blds[2][128 * 64];   //  2 x 16 KB
  __shared__ __align__(16) float  Cred[32 * 128];      // 16 KB
  __shared__ float norms[32];

  const int t    = threadIdx.x;
  const int lane = t & 31;
  const int wid  = t >> 5;
  const int m0   = blockIdx.x * 32;

  // wave sub-tile: 16 rows x 32 cols (2 WMMA accumulators)
  const int m_w  = 16 * (wid & 1);
  const int n_w  = 32 * (wid >> 1);
  const int half = (lane >> 4) * 8;   // ISA A/B frag: lanes 16-31 take k+8 span

  const v8f zf = {0.f, 0.f, 0.f, 0.f, 0.f, 0.f, 0.f, 0.f};
  v8f acc[2] = {zf, zf};

  // staging assignments
  const int arow = t >> 3;          // 0..31  (A: 32 rows x 64 k fp32)
  const int akq  = (t & 7) * 8;     // 0..56
  const int brow = t >> 1;          // 0..127 (B: 128 n x 64 k bf16, n-major)
  const int bseg = (t & 1) * 32;

  const float*  aptr = A  + (unsigned)(m0 + arow) * 8192u + akq;
  const __bf16* bptr = BT + (unsigned)brow * 8192u + bseg;
  __bf16* const adst = &Alds[0][arow * 64 + akq];   // buffer stride below
  __bf16* const bdst = &Blds[0][brow * 64 + bseg];
  const int ABUF = 32 * 64;    // elements between A buffers
  const int BBUF = 128 * 64;   // elements between B buffers

  const int NB = 128;  // 8192 / 64 K-chunks

  // ---- prologue: stage chunk 0 into buffer 0 ----
  {
    #pragma unroll
    for (int j = 0; j < 4; ++j)
      async_cp_b128(bdst + 8 * j, bptr + 8 * j);
    float4 fa0 = *(const float4*)(aptr);
    float4 fa1 = *(const float4*)(aptr + 4);
    *(v8bf*)(adst) = pack8(fa0, fa1);
    wait_async0();
  }
  __syncthreads();

  for (int kc = 0; kc < NB; ++kc) {
    const int cur = kc & 1;
    const int nxt = cur ^ 1;

    // ---- issue next chunk's loads before computing on current ----
    float4 pa0, pa1;
    if (kc + 1 < NB) {
      pa0 = *(const float4*)(aptr + (kc + 1) * 64);
      pa1 = *(const float4*)(aptr + (kc + 1) * 64 + 4);
      #pragma unroll
      for (int j = 0; j < 4; ++j)
        async_cp_b128(bdst + nxt * BBUF + 8 * j, bptr + (kc + 1) * 64 + 8 * j);
      if (kc + 2 < NB)
        __builtin_prefetch(aptr + (kc + 2) * 64, 0, 1);   // global_prefetch_b8
    }

    // ---- WMMA on current buffer ----
    const __bf16* Ab = &Alds[cur][0];
    const __bf16* Bb = &Blds[cur][0];
    #pragma unroll
    for (int ks = 0; ks < 64; ks += 32) {
      Frag fa;
      const __bf16* ap = Ab + (m_w + (lane & 15)) * 64 + ks + half;
      fa.h[0] = *(const v8bf*)(ap);
      fa.h[1] = *(const v8bf*)(ap + 16);
      #pragma unroll
      for (int nt = 0; nt < 2; ++nt) {
        Frag fb;
        const __bf16* bp = Bb + (n_w + nt * 16 + (lane & 15)) * 64 + ks + half;
        fb.h[0] = *(const v8bf*)(bp);
        fb.h[1] = *(const v8bf*)(bp + 16);
        acc[nt] = WMMA_BF16(fa.v, fb.v, acc[nt]);
      }
    }

    // ---- finish staging the next buffer, then sync ----
    if (kc + 1 < NB) {
      *(v8bf*)(adst + nxt * ABUF) = pack8(pa0, pa1);   // fp32->bf16 convert
      wait_async0();
    }
    __syncthreads();
  }

  // epilogue: relu into LDS (full 128-wide rows live in this WG)
  #pragma unroll
  for (int nt = 0; nt < 2; ++nt) {
    #pragma unroll
    for (int i = 0; i < 8; ++i) {
      int m = m_w + i + ((lane >> 4) << 3);
      int n = n_w + nt * 16 + (lane & 15);
      Cred[m * 128 + n] = fmaxf(acc[nt][i], 0.0f);
    }
  }
  __syncthreads();
  if (t < 32) {
    float s = 0.0f;
    #pragma unroll 4
    for (int n = 0; n < 128; ++n) { float v = Cred[t * 128 + n]; s += v * v; }
    norms[t] = do_norm ? (1.0f / fmaxf(sqrtf(s), 1e-12f)) : 1.0f;
  }
  __syncthreads();
  #pragma unroll
  for (int i = 0; i < 8; ++i) {
    int p = t + i * 256;          // 2048 bf16 pairs
    int m = p >> 6;
    int n = (p & 63) * 2;
    float inv = norms[m];
    __bf16* o = Xout + (unsigned)(m0 + m) * 128u + n;
    o[0] = (__bf16)(Cred[m * 128 + n]     * inv);
    o[1] = (__bf16)(Cred[m * 128 + n + 1] * inv);
  }
}

// ---------------------------------------------------------------------------
// YT (128x8192, n-major bf16) = (X[8192x128]bf16 @ W[128x128]fp32)^T
// WG tile = 128 rows x 128 cols; W transposed into LDS; output transposed
// through LDS so the big GEMM gets a contiguous-in-k B operand.
// ---------------------------------------------------------------------------
__global__ __launch_bounds__(256)
void k_gemm_xw(const __bf16* __restrict__ X, const float* __restrict__ W,
               __bf16* __restrict__ YT)
{
  __shared__ __align__(16) __bf16 Wlds[128 * 128];  // 32 KB (n-major)
  __shared__ __align__(16) __bf16 Tlds[128 * 128];  // 32 KB (transposed C)
  const int t    = threadIdx.x;
  const int lane = t & 31;
  const int wid  = t >> 5;
  const int m0   = blockIdx.x * 128;

  #pragma unroll
  for (int i = 0; i < 64; ++i) {
    int idx = t + i * 256;
    int k = idx >> 7, n = idx & 127;
    Wlds[n * 128 + k] = (__bf16)W[idx];
  }
  __syncthreads();

  const int m_w  = 32 * (wid & 3);
  const int n_w  = 64 * (wid >> 2);
  const int half = (lane >> 4) * 8;

  const v8f zf = {0.f, 0.f, 0.f, 0.f, 0.f, 0.f, 0.f, 0.f};
  v8f acc[8] = {zf, zf, zf, zf, zf, zf, zf, zf};

  #pragma unroll
  for (int ks = 0; ks < 128; ks += 32) {
    Frag fa[2];
    #pragma unroll
    for (int mt = 0; mt < 2; ++mt) {
      const __bf16* ap = X + (unsigned)(m0 + m_w + mt * 16 + (lane & 15)) * 128u + ks + half;
      fa[mt].h[0] = *(const v8bf*)(ap);
      fa[mt].h[1] = *(const v8bf*)(ap + 16);
    }
    #pragma unroll
    for (int nt = 0; nt < 4; ++nt) {
      Frag fb;
      const __bf16* bp = &Wlds[(n_w + nt * 16 + (lane & 15)) * 128 + ks + half];
      fb.h[0] = *(const v8bf*)(bp);
      fb.h[1] = *(const v8bf*)(bp + 16);
      #pragma unroll
      for (int mt = 0; mt < 2; ++mt)
        acc[mt * 4 + nt] = WMMA_BF16(fa[mt].v, fb.v, acc[mt * 4 + nt]);
    }
  }

  #pragma unroll
  for (int mt = 0; mt < 2; ++mt)
    #pragma unroll
    for (int nt = 0; nt < 4; ++nt)
      #pragma unroll
      for (int i = 0; i < 8; ++i) {
        int m = m_w + mt * 16 + i + ((lane >> 4) << 3);
        int n = n_w + nt * 16 + (lane & 15);
        Tlds[n * 128 + m] = (__bf16)acc[mt * 4 + nt][i];
      }
  __syncthreads();

  const int n = t >> 1, seg = (t & 1) * 64;
  #pragma unroll
  for (int j = 0; j < 8; ++j)
    *(v8bf*)(YT + (unsigned)n * 8192u + m0 + seg + 8 * j) =
        *(const v8bf*)&Tlds[n * 128 + seg + 8 * j];
}

// ---------------------------------------------------------------------------
// score[m] += MLP(x[m]) : relu(x@Wa+ba) -> relu(h@Wb+bb) -> @Wc+bc
// Weights pre-transposed bf16 in global (L2-resident; ~192KB total).
// WG = 32 rows; both layers via WMMA; final dot fused into the h2 epilogue.
// ---------------------------------------------------------------------------
__global__ __launch_bounds__(256)
void k_mlp(const __bf16* __restrict__ X, const __bf16* __restrict__ WaT,
           const __bf16* __restrict__ WbT, const float* __restrict__ ba,
           const float* __restrict__ bb, const float* __restrict__ Wc,
           const float* __restrict__ bc, float* __restrict__ score)
{
  __shared__ __align__(16) __bf16 H1[32 * 256];   // 16 KB
  __shared__ __align__(16) float  H2[32 * 256];   // 32 KB
  __shared__ float bal[256], bbl[256], wcl[256];

  const int t    = threadIdx.x;
  const int lane = t & 31;
  const int wid  = t >> 5;
  const int m0   = blockIdx.x * 32;

  bal[t] = ba[t]; bbl[t] = bb[t]; wcl[t] = Wc[t];

  const int m_w  = 16 * (wid & 1);
  const int n_w  = 64 * (wid >> 1);
  const int half = (lane >> 4) * 8;

  const v8f zf = {0.f, 0.f, 0.f, 0.f, 0.f, 0.f, 0.f, 0.f};
  v8f acc1[4] = {zf, zf, zf, zf};

  // layer a: [32x128] @ [128x256]
  #pragma unroll
  for (int ks = 0; ks < 128; ks += 32) {
    Frag fa;
    const __bf16* ap = X + (unsigned)(m0 + m_w + (lane & 15)) * 128u + ks + half;
    fa.h[0] = *(const v8bf*)(ap);
    fa.h[1] = *(const v8bf*)(ap + 16);
    #pragma unroll
    for (int nt = 0; nt < 4; ++nt) {
      Frag fb;
      const __bf16* bp = WaT + (unsigned)(n_w + nt * 16 + (lane & 15)) * 128u + ks + half;
      fb.h[0] = *(const v8bf*)(bp);
      fb.h[1] = *(const v8bf*)(bp + 16);
      acc1[nt] = WMMA_BF16(fa.v, fb.v, acc1[nt]);
    }
  }
  __syncthreads();   // bal ready + H1 region free
  #pragma unroll
  for (int nt = 0; nt < 4; ++nt)
    #pragma unroll
    for (int i = 0; i < 8; ++i) {
      int m = m_w + i + ((lane >> 4) << 3);
      int n = n_w + nt * 16 + (lane & 15);
      H1[m * 256 + n] = (__bf16)fmaxf(acc1[nt][i] + bal[n], 0.0f);
    }
  __syncthreads();

  // layer b: [32x256] @ [256x256]
  v8f acc2[4] = {zf, zf, zf, zf};
  #pragma unroll
  for (int ks = 0; ks < 256; ks += 32) {
    Frag fa;
    const __bf16* ap = &H1[(m_w + (lane & 15)) * 256 + ks + half];
    fa.h[0] = *(const v8bf*)(ap);
    fa.h[1] = *(const v8bf*)(ap + 16);
    #pragma unroll
    for (int nt = 0; nt < 4; ++nt) {
      Frag fb;
      const __bf16* bp = WbT + (unsigned)(n_w + nt * 16 + (lane & 15)) * 256u + ks + half;
      fb.h[0] = *(const v8bf*)(bp);
      fb.h[1] = *(const v8bf*)(bp + 16);
      acc2[nt] = WMMA_BF16(fa.v, fb.v, acc2[nt]);
    }
  }
  #pragma unroll
  for (int nt = 0; nt < 4; ++nt)
    #pragma unroll
    for (int i = 0; i < 8; ++i) {
      int m = m_w + i + ((lane >> 4) << 3);
      int n = n_w + nt * 16 + (lane & 15);
      H2[m * 256 + n] = fmaxf(acc2[nt][i] + bbl[n], 0.0f) * wcl[n];
    }
  __syncthreads();
  if (t < 32) {
    float s = bc[0];
    #pragma unroll 4
    for (int n = 0; n < 256; ++n) s += H2[t * 256 + n];
    score[m0 + t] += s;
  }
}

// ---------------------------------------------------------------------------
// one-time converts + init
// ---------------------------------------------------------------------------
__global__ __launch_bounds__(256)
void k_cvt_w1t(const float* __restrict__ W1, __bf16* __restrict__ W1T) {
  int idx = blockIdx.x * 256 + threadIdx.x;     // 1,048,576
  int k = idx >> 7, n = idx & 127;
  W1T[(unsigned)n * 8192u + k] = (__bf16)W1[idx];
}

__global__ __launch_bounds__(256)
void k_cvt_mlp(const float* __restrict__ Wa, const float* __restrict__ Wb,
               __bf16* __restrict__ WaT, __bf16* __restrict__ WbT) {
  int idx = blockIdx.x * 256 + threadIdx.x;     // 98,304
  if (idx < 32768) {
    int k = idx >> 8, n = idx & 255;
    WaT[n * 128 + k] = (__bf16)Wa[idx];
  } else {
    int j = idx - 32768;
    int k = j >> 8, n = j & 255;
    WbT[n * 256 + k] = (__bf16)Wb[j];
  }
}

__global__ __launch_bounds__(256)
void k_zero(float* __restrict__ out) { out[blockIdx.x * 256 + threadIdx.x] = 0.0f; }

// ---------------------------------------------------------------------------
extern "C" void kernel_launch(void* const* d_in, const int* in_sizes, int n_in,
                              void* d_out, int out_size, void* d_ws, size_t ws_size,
                              hipStream_t stream)
{
  (void)in_sizes; (void)n_in; (void)out_size; (void)ws_size;
  const float* adj1 = (const float*)d_in[0];
  const float* adj2 = (const float*)d_in[1];
  const float* W1   = (const float*)d_in[2];
  const float* Wl[6] = { (const float*)d_in[3], (const float*)d_in[4],
                         (const float*)d_in[5], (const float*)d_in[6],
                         (const float*)d_in[7], (const float*)d_in[8] };
  const float* Wa = (const float*)d_in[9];
  const float* ba = (const float*)d_in[10];
  const float* Wb = (const float*)d_in[11];
  const float* bb = (const float*)d_in[12];
  const float* Wc = (const float*)d_in[13];
  const float* bc = (const float*)d_in[14];
  float* score = (float*)d_out;

  char* w = (char*)d_ws;
  __bf16* x    = (__bf16*)(w);                                  // 2 MB
  __bf16* yT   = (__bf16*)(w + (size_t)(2 << 20));              // 2 MB (128x8192)
  __bf16* W1T  = (__bf16*)(w + (size_t)(4 << 20));              // 2 MB (128x8192)
  __bf16* WaT  = (__bf16*)(w + (size_t)(6 << 20));              // 64 KB
  __bf16* WbT  = (__bf16*)(w + (size_t)(6 << 20) + (64 << 10)); // 128 KB

  k_zero   <<<32,   256, 0, stream>>>(score);
  k_cvt_w1t<<<4096, 256, 0, stream>>>(W1, W1T);
  k_cvt_mlp<<<384,  256, 0, stream>>>(Wa, Wb, WaT, WbT);

  // layer 1: x = l2norm(relu(adj1 @ W1))
  k_gemm_adj<<<256, 256, 0, stream>>>(adj1, W1T, x, 1);
  k_mlp     <<<256, 256, 0, stream>>>(x, WaT, WbT, ba, bb, Wc, bc, score);

  // layers 2..7
  for (int i = 0; i < 6; ++i) {
    k_gemm_xw <<<64,  256, 0, stream>>>(x, Wl[i], yT);
    k_gemm_adj<<<256, 256, 0, stream>>>(adj2, yT, x, i < 5 ? 1 : 0);
    k_mlp     <<<256, 256, 0, stream>>>(x, WaT, WbT, ba, bb, Wc, bc, score);
  }
}
